// GSPN_39779987095925
// MI455X (gfx1250) — compile-verified, matrix operand order checked by reading, not run on hardware
//
#include <hip/hip_runtime.h>

// ---------------------------------------------------------------------------
// Types for CDNA5 WMMA (wave32, gfx1250)
// ---------------------------------------------------------------------------
typedef __bf16 bf16;
typedef __attribute__((ext_vector_type(16))) __bf16 v16bf;
typedef __attribute__((ext_vector_type(8)))  __bf16 v8bf;
typedef __attribute__((ext_vector_type(8)))  float  v8f;

#define BD 4
#define TD 1024
#define DD 768
#define HWD 32
#define NP 4096          // B*T pixels
#define DSD 48

__device__ __forceinline__ float sigm(float x) { return 1.0f / (1.0f + __expf(-x)); }

// ---------------------------------------------------------------------------
// Kernel 1: LayerNorm over D per token -> bf16 pixel-major [p][768]
//           + shortcut copy into d_out second half
// ---------------------------------------------------------------------------
__global__ void k_ln(const float* __restrict__ hs,
                     const float* __restrict__ nw,
                     const float* __restrict__ nb,
                     bf16* __restrict__ xln,
                     float* __restrict__ shortcut)
{
    const int p   = blockIdx.x;            // 0..4095
    const int tid = threadIdx.x;           // 0..255
    const float* row = hs + (size_t)p * DD;

    float v0 = row[tid];
    float v1 = row[tid + 256];
    float v2 = row[tid + 512];
    shortcut[(size_t)p * DD + tid      ] = v0;
    shortcut[(size_t)p * DD + tid + 256] = v1;
    shortcut[(size_t)p * DD + tid + 512] = v2;

    float s = v0 + v1 + v2;
    float q = v0 * v0 + v1 * v1 + v2 * v2;
    for (int o = 16; o > 0; o >>= 1) {
        s += __shfl_down(s, o, 32);
        q += __shfl_down(q, o, 32);
    }
    __shared__ float red[16];
    const int wave = tid >> 5, lane = tid & 31;
    if (lane == 0) { red[wave] = s; red[8 + wave] = q; }
    __syncthreads();
    if (tid == 0) {
        float S = 0.f, Q = 0.f;
        for (int i = 0; i < 8; ++i) { S += red[i]; Q += red[8 + i]; }
        float mean = S * (1.0f / DD);
        float var  = Q * (1.0f / DD) - mean * mean;
        red[0] = mean;
        red[1] = rsqrtf(var + 1e-5f);
    }
    __syncthreads();
    const float mean = red[0], rstd = red[1];

    bf16* o = xln + (size_t)p * DD;
    o[tid      ] = (bf16)((v0 - mean) * rstd * nw[tid]       + nb[tid]);
    o[tid + 256] = (bf16)((v1 - mean) * rstd * nw[tid + 256] + nb[tid + 256]);
    o[tid + 512] = (bf16)((v2 - mean) * rstd * nw[tid + 512] + nb[tid + 512]);
}

// ---------------------------------------------------------------------------
// Kernel 2: bf16 WMMA GEMM, double-buffered LDS, one barrier per K-step.
// C[M][N] = A[M][K] * B[K][N] with B given transposed (Bt[N][K], pixel-major).
// Block: 256 thr = 8 waves arranged 2(M) x 4(N); block tile 64 x 256;
// wave tile 32 x 64 = 2x4 v_wmma_f32_16x16x32_bf16 fragments; BK = 32.
// TRANS_STORE: write C transposed as [N][M] (for the final out-proj).
// ---------------------------------------------------------------------------
#define LDSW 40                 // padded K-stride (bf16) => conflict-free b128 reads
#define ASZ  (64 * LDSW)        // A tile elems per buffer
#define BSZ  (256 * LDSW)       // B tile elems per buffer

__device__ __forceinline__ v16bf load_frag(const bf16* rowp, int half)
{
    const v8bf lo = *(const v8bf*)(rowp + half * 8);
    const v8bf hi = *(const v8bf*)(rowp + 16 + half * 8);
    return __builtin_shufflevector(lo, hi, 0, 1, 2, 3, 4, 5, 6, 7,
                                           8, 9, 10, 11, 12, 13, 14, 15);
}

__device__ __forceinline__ void stage_tile(const bf16* __restrict__ A,
                                           const bf16* __restrict__ Bt,
                                           bf16* sAb, bf16* sBb,
                                           int bm0, int bn0, int k0,
                                           int M, int K, int tid)
{
    // A tile: 64 rows x 32 bf16 (zero-padded past M) -- one b128 per thread
    {
        const int row = tid >> 2, c4 = tid & 3;
        const int gm = bm0 + row;
        v8bf v;
        if (gm < M) v = *(const v8bf*)(A + (size_t)gm * K + k0 + c4 * 8);
        else { bf16 z = (bf16)0.0f; v = (v8bf){z, z, z, z, z, z, z, z}; }
        *(v8bf*)&sAb[row * LDSW + c4 * 8] = v;
    }
    // B tile: 256 rows x 32 bf16 -- four b128 per thread
    #pragma unroll
    for (int r = 0; r < 4; ++r) {
        const int idx = tid + r * 256;
        const int row = idx >> 2, c4 = idx & 3;
        *(v8bf*)&sBb[row * LDSW + c4 * 8] =
            *(const v8bf*)(Bt + (size_t)(bn0 + row) * K + k0 + c4 * 8);
    }
}

template <bool TRANS_STORE>
__global__ void k_gemm(const bf16* __restrict__ A,   // [M][K]
                       const bf16* __restrict__ Bt,  // [N][K]
                       float* __restrict__ C,
                       int M, int N, int K)
{
    __shared__ __align__(16) bf16 sA[2 * ASZ];
    __shared__ __align__(16) bf16 sB[2 * BSZ];

    const int tid  = threadIdx.x;
    const int bn0  = blockIdx.x * 256;
    const int bm0  = blockIdx.y * 64;
    const int wave = tid >> 5;
    const int wm   = wave >> 2;        // 0..1
    const int wn   = wave & 3;         // 0..3
    const int lane = tid & 31;
    const int half = lane >> 4;
    const int rrow = lane & 15;

    v8f acc[2][4];
    #pragma unroll
    for (int mf = 0; mf < 2; ++mf)
        #pragma unroll
        for (int nf = 0; nf < 4; ++nf) acc[mf][nf] = {};

    const int KT = K / 32;
    stage_tile(A, Bt, sA, sB, bm0, bn0, 0, M, K, tid);

    for (int kt = 0; kt < KT; ++kt) {
        __syncthreads();
        const int cur = kt & 1;
        if (kt + 1 < KT)
            stage_tile(A, Bt, sA + (cur ^ 1) * ASZ, sB + (cur ^ 1) * BSZ,
                       bm0, bn0, (kt + 1) * 32, M, K, tid);
        if (kt + 2 < KT) {              // global_prefetch_b8 two tiles ahead
            int gm = bm0 + (tid >> 2); if (gm >= M) gm = M - 1;
            __builtin_prefetch(A  + (size_t)gm * K + (kt + 2) * 32, 0, 1);
            __builtin_prefetch(Bt + (size_t)(bn0 + (tid >> 2) * 4) * K + (kt + 2) * 32, 0, 1);
        }

        const bf16* cA = sA + cur * ASZ;
        const bf16* cB = sB + cur * BSZ;

        v16bf af[2];
        #pragma unroll
        for (int mf = 0; mf < 2; ++mf)
            af[mf] = load_frag(&cA[(wm * 32 + mf * 16 + rrow) * LDSW], half);

        #pragma unroll
        for (int nf = 0; nf < 4; ++nf) {
            const v16bf bfm = load_frag(&cB[(wn * 64 + nf * 16 + rrow) * LDSW], half);
            #pragma unroll
            for (int mf = 0; mf < 2; ++mf)
                acc[mf][nf] = __builtin_amdgcn_wmma_f32_16x16x32_bf16(
                    false, af[mf], false, bfm, (short)0, acc[mf][nf],
                    false, false);
        }
    }

    // Store.  C frag layout: lane -> N = rrow, VGPR r -> M = half*8 + r.
    #pragma unroll
    for (int mf = 0; mf < 2; ++mf) {
        #pragma unroll
        for (int nf = 0; nf < 4; ++nf) {
            const int n     = bn0 + wn * 64 + nf * 16 + rrow;
            const int mbase = bm0 + wm * 32 + mf * 16 + half * 8;
            if (TRANS_STORE) {
                float* dst = C + (size_t)n * M + mbase;   // 8 contiguous floats
                #pragma unroll
                for (int r = 0; r < 8; ++r) dst[r] = acc[mf][nf][r];
            } else {
                #pragma unroll
                for (int r = 0; r < 8; ++r)
                    if (mbase + r < M)
                        C[(size_t)(mbase + r) * N + n] = acc[mf][nf][r];
            }
        }
    }
}

// ---------------------------------------------------------------------------
// Kernel 3: depthwise conv (2R+1)^2, data layout [D][NP], p = b*1024 + h*32+w.
// Optional bias; optional x*relu(x) activation on output.
// ---------------------------------------------------------------------------
template <int R, bool SQRELU>
__global__ void k_dwconv(const float* __restrict__ in,
                         const float* __restrict__ w,
                         const float* __restrict__ bias,
                         float* __restrict__ out)
{
    const int b = blockIdx.x / DD;
    const int d = blockIdx.x % DD;
    const size_t base = (size_t)d * NP + (size_t)b * TD;

    __shared__ float t[32 * 33];
    for (int idx = threadIdx.x; idx < 1024; idx += 256) {
        t[(idx >> 5) * 33 + (idx & 31)] = in[base + idx];
    }
    __syncthreads();

    const int KW = 2 * R + 1;
    const float* wp = w + (size_t)d * KW * KW;
    const float bv = bias ? bias[d] : 0.0f;

    for (int idx = threadIdx.x; idx < 1024; idx += 256) {
        const int hh = idx >> 5, ww = idx & 31;
        float acc = bv;
        #pragma unroll
        for (int i = 0; i < KW; ++i) {
            const int y0 = hh + i - R;
            #pragma unroll
            for (int j = 0; j < KW; ++j) {
                const int x0 = ww + j - R;
                const float v = (y0 >= 0 && y0 < 32 && x0 >= 0 && x0 < 32)
                                    ? t[y0 * 33 + x0] : 0.0f;
                acc = fmaf(wp[i * KW + j], v, acc);
            }
        }
        if (SQRELU) acc = acc * fmaxf(acc, 0.0f);
        out[base + idx] = acc;
    }
}

// ---------------------------------------------------------------------------
// Kernel 4: pack f32 d-major [768][4096] -> bf16 pixel-major [4096][768]
// ---------------------------------------------------------------------------
__global__ void k_pack(const float* __restrict__ in, bf16* __restrict__ out)
{
    __shared__ float t[32][33];
    const int p0 = blockIdx.x * 32;
    const int d0 = blockIdx.y * 32;
    const int tx = threadIdx.x, ty = threadIdx.y;     // 32 x 8
    #pragma unroll
    for (int i = 0; i < 32; i += 8)
        t[ty + i][tx] = in[(size_t)(d0 + ty + i) * NP + p0 + tx];
    __syncthreads();
    #pragma unroll
    for (int i = 0; i < 32; i += 8)
        out[(size_t)(p0 + ty + i) * DD + d0 + tx] = (bf16)t[tx][ty + i];
}

// ---------------------------------------------------------------------------
// Kernel 5: fused 4-direction GSPN scan.
// One wave per (b, d, nc), lane = h (wave32 == H=32).  Gate/L/U/D projections
// computed as 48-dot products against an LDS tile of xp; h+/-1 coupling via
// ds_bpermute; accumulates sum_k m_w[k] * (h*U + x*D) in registers.
// ---------------------------------------------------------------------------
__global__ void __launch_bounds__(256, 2)
k_scan(const float* __restrict__ xp,    // [48][4096]
       const float* __restrict__ x2,    // [768][4096]
       const float* __restrict__ wup,   // [9216][48]
       const float* __restrict__ lup,   // [3072][48]
       const float* __restrict__ uup,   // [3072][48]
       const float* __restrict__ dco,   // [3072][48]
       const float* __restrict__ mw,    // [4]
       float* __restrict__ y)           // [768][4096]
{
    const int bn = blockIdx.y;           // b*4 + nc
    const int b  = bn >> 2, nc = bn & 3;
    const int wave = threadIdx.x >> 5;
    const int lane = threadIdx.x & 31;   // = h
    const int d = blockIdx.x * 8 + wave;
    const int h = lane;
    const int basep = b * TD + nc * 8;

    __shared__ float sxp[48 * 256];      // [j][h][s], 48 KB
    for (int idx = threadIdx.x; idx < 48 * 256; idx += 256) {
        const int j = idx >> 8, rem = idx & 255;
        const int hh = rem >> 3, s = rem & 7;
        sxp[idx] = xp[(size_t)j * NP + basep + hh * 32 + s];
    }
    __syncthreads();

    float y8[8] = {0.f, 0.f, 0.f, 0.f, 0.f, 0.f, 0.f, 0.f};

    for (int k = 0; k < 4; ++k) {
        const int cu = __builtin_amdgcn_readfirstlane(k * DD + d);
        const float* wgl = wup + (size_t)cu * DSD;
        const float* wgm = wup + (size_t)(3072 + cu) * DSD;
        const float* wgr = wup + (size_t)(6144 + cu) * DSD;
        const float* wl  = lup + (size_t)cu * DSD;
        const float* wu_ = uup + (size_t)cu * DSD;
        const float* wd_ = dco + (size_t)cu * DSD;

        float agl[8] = {}, agm[8] = {}, agr[8] = {};
        float al [8] = {}, au [8] = {}, ad [8] = {};

        for (int j = 0; j < DSD; ++j) {
            const float* xr = &sxp[j * 256 + h * 8];
            const float cgl = wgl[j], cgm = wgm[j], cgr = wgr[j];
            const float cl = wl[j], cu_ = wu_[j], cd = wd_[j];
            #pragma unroll
            for (int s = 0; s < 8; ++s) {
                const float xv = xr[s];
                agl[s] = fmaf(cgl, xv, agl[s]);
                agm[s] = fmaf(cgm, xv, agm[s]);
                agr[s] = fmaf(cgr, xv, agr[s]);
                al [s] = fmaf(cl,  xv, al [s]);
                au [s] = fmaf(cu_, xv, au [s]);
                ad [s] = fmaf(cd,  xv, ad [s]);
            }
        }

        const float mk = mw[k];
        const float* x2b = x2 + (size_t)d * NP + b * TD;
        float hcur = 0.0f;
        #pragma unroll
        for (int s = 0; s < 8; ++s) {
            const int w = nc * 8 + s;
            int h2, w2;
            if      (k == 0) { h2 = h;      w2 = w;      }
            else if (k == 1) { h2 = w;      w2 = h;      }
            else if (k == 2) { h2 = h;      w2 = 31 - w; }
            else             { h2 = 31 - w; w2 = h;      }
            const float xv = x2b[h2 * 32 + w2];

            float gl = sigm(agl[s]);
            float gm = sigm(agm[s]);
            float gr = sigm(agr[s]);
            float ss = (h == 0) ? (gm + gr) : ((h == 31) ? (gl + gm) : (gl + gm + gr));
            ss = fmaxf(ss, 1e-7f);
            const float inv = 1.0f / ss;
            gl *= inv; gm *= inv; gr *= inv;

            const int vu = __builtin_amdgcn_ds_bpermute(((lane + 31) & 31) << 2,
                                                        __float_as_int(hcur));
            const int vd = __builtin_amdgcn_ds_bpermute(((lane + 1) & 31) << 2,
                                                        __float_as_int(hcur));
            const float hup = (h == 0)  ? 0.0f : __int_as_float(vu);
            const float hdn = (h == 31) ? 0.0f : __int_as_float(vd);

            hcur = al[s] * xv + gl * hup + gm * hcur + gr * hdn;
            y8[s] = fmaf(mk, hcur * au[s] + xv * ad[s], y8[s]);
        }
    }

    float* yp = y + (size_t)d * NP + b * TD + h * 32 + nc * 8;
    #pragma unroll
    for (int s = 0; s < 8; ++s) yp[s] = y8[s];
}

// ---------------------------------------------------------------------------
// Kernel 6: f32 -> bf16 weight conversion
// ---------------------------------------------------------------------------
__global__ void k_cvt(const float* __restrict__ src, bf16* __restrict__ dst, int n)
{
    const int i = blockIdx.x * 256 + threadIdx.x;
    if (i < n) dst[i] = (bf16)src[i];
}

// ---------------------------------------------------------------------------
// Orchestration
// ---------------------------------------------------------------------------
extern "C" void kernel_launch(void* const* d_in, const int* in_sizes, int n_in,
                              void* d_out, int out_size, void* d_ws, size_t ws_size,
                              hipStream_t stream)
{
    (void)in_sizes; (void)n_in; (void)out_size; (void)ws_size;

    const float* hs      = (const float*)d_in[0];
    const float* norm_w  = (const float*)d_in[1];
    const float* norm_b  = (const float*)d_in[2];
    const float* in_proj = (const float*)d_in[3];
    const float* conv7w  = (const float*)d_in[4];
    const float* conv7b  = (const float*)d_in[5];
    const float* xdown   = (const float*)d_in[6];
    const float* wup     = (const float*)d_in[7];
    const float* lup     = (const float*)d_in[8];
    const float* uup     = (const float*)d_in[9];
    const float* dco     = (const float*)d_in[10];
    const float* mwv     = (const float*)d_in[11];
    const float* outconv = (const float*)d_in[12];
    const float* outdcw  = (const float*)d_in[13];
    const float* outproj = (const float*)d_in[14];

    float* out = (float*)d_out;                    // [BTD out][BTD shortcut]
    char*  ws  = (char*)d_ws;

    const size_t SZ_DM = (size_t)DD * NP * 4;      // 12.58 MB f32 d-major
    const size_t SZ_PM = (size_t)NP * DD * 2;      //  6.29 MB bf16 pixel-major
    float* bufA   = (float*)(ws);                  // x1, later z1
    float* bufB   = (float*)(ws + SZ_DM);          // x2, later z2r
    float* bufY   = (float*)(ws + 2 * SZ_DM);      // y
    bf16*  pm1    = (bf16*) (ws + 3 * SZ_DM);                // xln -> y_pm -> z2r_pm
    bf16*  pm2    = (bf16*) (ws + 3 * SZ_DM + SZ_PM);        // x2_pm
    float* xpbuf  = (float*)(ws + 3 * SZ_DM + 2 * SZ_PM);    // [48][4096]
    char*  wbase  =         ws + 3 * SZ_DM + 2 * SZ_PM + (size_t)DSD * NP * 4;
    bf16* w_ip = (bf16*)(wbase);
    bf16* w_oc = (bf16*)(wbase + (size_t)DD * DD * 2);
    bf16* w_op = (bf16*)(wbase + (size_t)DD * DD * 4);
    bf16* w_xd = (bf16*)(wbase + (size_t)DD * DD * 6);

    // Weight conversions to bf16
    k_cvt<<<(DD * DD + 255) / 256, 256, 0, stream>>>(in_proj, w_ip, DD * DD);
    k_cvt<<<(DD * DD + 255) / 256, 256, 0, stream>>>(outconv, w_oc, DD * DD);
    k_cvt<<<(DD * DD + 255) / 256, 256, 0, stream>>>(outproj, w_op, DD * DD);
    k_cvt<<<(DSD * DD + 255) / 256, 256, 0, stream>>>(xdown,  w_xd, DSD * DD);

    // 1. LayerNorm + shortcut
    k_ln<<<NP, 256, 0, stream>>>(hs, norm_w, norm_b, pm1, out + (size_t)BD * TD * DD);

    // 2. in_proj GEMM: x1[768][4096] = W_ip * xln
    k_gemm<false><<<dim3(NP / 256, DD / 64), 256, 0, stream>>>(w_ip, pm1, bufA, DD, NP, DD);

    // 3. depthwise 7x7 (+bias) -> x2
    k_dwconv<3, false><<<BD * DD, 256, 0, stream>>>(bufA, conv7w, conv7b, bufB);

    // 4. pack x2 -> bf16 pm
    k_pack<<<dim3(NP / 32, DD / 32), dim3(32, 8), 0, stream>>>(bufB, pm2);

    // 5. xdown GEMM: xp[48][4096]
    k_gemm<false><<<dim3(NP / 256, 1), 256, 0, stream>>>(w_xd, pm2, xpbuf, DSD, NP, DD);

    // 6. fused 4-direction GSPN scan -> y[768][4096]
    k_scan<<<dim3(DD / 8, BD * 4), 256, 0, stream>>>(xpbuf, bufB, wup, lup, uup, dco, mwv, bufY);

    // 7. pack y -> bf16 pm
    k_pack<<<dim3(NP / 32, DD / 32), dim3(32, 8), 0, stream>>>(bufY, pm1);

    // 8. outconv GEMM: z1[768][4096]
    k_gemm<false><<<dim3(NP / 256, DD / 64), 256, 0, stream>>>(w_oc, pm1, bufA, DD, NP, DD);

    // 9. depthwise 3x3 + x*relu(x) -> z2r
    k_dwconv<1, true><<<BD * DD, 256, 0, stream>>>(bufA, outdcw, nullptr, bufB);

    // 10. pack z2r -> bf16 pm
    k_pack<<<dim3(NP / 32, DD / 32), dim3(32, 8), 0, stream>>>(bufB, pm1);

    // 11. outproj GEMM with transposed store directly into d_out [p][768]
    k_gemm<true><<<dim3(NP / 256, DD / 64), 256, 0, stream>>>(w_op, pm1, out, DD, NP, DD);
}
